// GATRelateCNet_85023172591808
// MI455X (gfx1250) — compile-verified, matrix operand order checked by reading, not run on hardware
//
#include <hip/hip_runtime.h>

// ===========================================================================
// GATRelateCNet forward for MI455X (gfx1250, wave32, WMMA).
//
// Conv1x3 layers are LDS-staged implicit GEMMs executed with
// v_wmma_f32_16x16x32_f16 (f16 inputs, f32 accumulate). The K dimension is
// repacked as k = ci*4 + t (t=3 tap zero-padded) so that:
//   - A (weights) fragments are 2 x ds_load_b128 per lane (16B aligned)
//   - B (input) fragments are 8 x aligned 32-bit LDS loads of packed half2
// Everything else (LN, GCN scatter/normalize, pool, upsample, BN) is
// bandwidth-bound elementwise/reduction work in fp32.
//
// Input leaf order (setup_inputs dict, insertion order, params flattened):
//   0 x | 1 stem_w | 2 stem_b | 3 stem_ln_w | 4 stem_ln_b | 5 scale
//   6+12*blk: conv_w[0..2], conv_b[0..2], ln_w[0..2], ln_b[0..2]  (blk=0..2)
//   42..44 up_w[0..2] | 45..47 up_b | 48..50 bn_g | 51..53 bn_b
//   54 out_w | 55 out_b | 56 edge_index (int32, 2xE) | 57 batch (unused)
// ===========================================================================

typedef __attribute__((ext_vector_type(16))) _Float16 v16h;
typedef __attribute__((ext_vector_type(8)))  float    v8f;

#define LN_EPS_F 1e-5f
#define BN_EPS_F 1e-5f

// ---------------------------------------------------------------- utilities
__global__ void k_zero(float* __restrict__ p, long n) {
  for (long i = (long)blockIdx.x * blockDim.x + threadIdx.x; i < n;
       i += (long)gridDim.x * blockDim.x)
    p[i] = 0.f;
}

// degree: deg[col] += 1 over edges
__global__ void k_deg(const int* __restrict__ ei, float* __restrict__ deg, int E) {
  for (long e = (long)blockIdx.x * blockDim.x + threadIdx.x; e < E;
       e += (long)gridDim.x * blockDim.x)
    atomicAdd(&deg[ei[E + e]], 1.0f);
}

// dinv = deg>0 ? rsqrt(max(deg,1)) : 0   (in-place)
__global__ void k_dinv(float* __restrict__ d, int n) {
  for (long i = (long)blockIdx.x * blockDim.x + threadIdx.x; i < n;
       i += (long)gridDim.x * blockDim.x) {
    float v = d[i];
    d[i] = (v > 0.f) ? rsqrtf(fmaxf(v, 1.f)) : 0.f;
  }
}

// agg[col] += dinv[row]*dinv[col]*x[row]  (edges only touch nodes < 2400;
// 2400*128*4B = 1.2MB -> L2 resident, fp32 atomics are cheap here)
__global__ void k_gcn_scatter(const float* __restrict__ x, const int* __restrict__ ei,
                              const float* __restrict__ dinv, float* __restrict__ agg,
                              int E, int S) {
  long total = (long)E * S;
  for (long i = (long)blockIdx.x * blockDim.x + threadIdx.x; i < total;
       i += (long)gridDim.x * blockDim.x) {
    long e = i / S;
    int  j = (int)(i - e * S);
    int  r = ei[e], c = ei[E + e];
    float en = dinv[r] * dinv[c];
    atomicAdd(&agg[(long)c * S + j], en * x[(long)r * S + j]);
  }
}

// out[node] = scale * sqrt(|x|^2) / sqrt(max(|agg|^2,1e-24)) * agg[node]
// One wave32 per node; in-place on x (row fully read before written).
__global__ __launch_bounds__(256)
void k_gcn_finalize(float* __restrict__ x, const float* __restrict__ agg,
                    const float* __restrict__ scalep, int S, int nActive) {
  int wave = threadIdx.x >> 5;
  int lane = threadIdx.x & 31;
  int node = blockIdx.x * (blockDim.x >> 5) + wave;
  if (node >= nActive) return;
  float ss = 0.f, xx = 0.f;
  for (int j = lane; j < S; j += 32) {
    float av = agg[(long)node * S + j];
    float xv = x[(long)node * S + j];
    ss += av * av;
    xx += xv * xv;
  }
  for (int o = 16; o > 0; o >>= 1) {
    ss += __shfl_xor(ss, o, 32);
    xx += __shfl_xor(xx, o, 32);
  }
  float r = scalep[0] * sqrtf(xx) * rsqrtf(fmaxf(ss, 1e-24f));
  for (int j = lane; j < S; j += 32)
    x[(long)node * S + j] = r * agg[(long)node * S + j];
}

// LayerNorm over a contiguous group (P*W per (b,c)); optional residual add.
// In-place on y (per-thread index sets are disjoint).
__global__ __launch_bounds__(256)
void k_layernorm(float* __restrict__ y, const float* __restrict__ lw,
                 const float* __restrict__ lb, const float* __restrict__ res,
                 int gs) {
  long base = (long)blockIdx.x * gs;
  __shared__ float s1[256], s2[256];
  float a = 0.f, q = 0.f;
  for (int i = threadIdx.x; i < gs; i += blockDim.x) {
    float v = y[base + i];
    a += v;
    q += v * v;
  }
  s1[threadIdx.x] = a;
  s2[threadIdx.x] = q;
  __syncthreads();
  for (int s = 128; s > 0; s >>= 1) {
    if ((int)threadIdx.x < s) {
      s1[threadIdx.x] += s1[threadIdx.x + s];
      s2[threadIdx.x] += s2[threadIdx.x + s];
    }
    __syncthreads();
  }
  float mu   = s1[0] / (float)gs;
  float var  = s2[0] / (float)gs - mu * mu;
  float rstd = rsqrtf(var + LN_EPS_F);
  for (int i = threadIdx.x; i < gs; i += blockDim.x) {
    float v = (y[base + i] - mu) * rstd * lw[i] + lb[i];
    if (res) v += res[base + i];
    y[base + i] = v;
  }
}

// ------------------------------------------------- conv1x3 as WMMA GEMM
// One block (8 waves) per (b,p) row. Input row (+halo, zero pad) and the
// full weight matrix are staged in LDS once; each wave then computes
// 16(cout) x 16(width) fp32 tiles with v_wmma_f32_16x16x32_f16.
//
// K packing: k = ci*4 + t, t in 0..3, weight(t=3) staged as 0.
//   A 16x32 f16 : lane%16 = M(co), VGPR v -> k0 = (v/4)*16 + (lane/16)*8 + (v%4)*2
//                 -> per lane two contiguous 16B runs => 2 x ds_load_b128
//   B 32x16 f16 : lane%16 = N(w); ldsX[ci][j] packs (Xpad(j), Xpad(j+1)) in
//                 one 32-bit word => each element pair is one aligned b32 load
//   D 16x16 f32 : lane%16 = N, VGPR r -> M = r + 8*(lane/16)
__global__ __launch_bounds__(256)
void k_conv1x3_wmma(const float* __restrict__ in, const float* __restrict__ wgt,
                    const float* __restrict__ bias, float* __restrict__ out,
                    int Cin, int Cout, int P, int W,
                    int Kpad32, int CinP, int CoutP) {
  extern __shared__ char smem[];
  _Float16* ldsW = (_Float16*)smem;                                 // CoutP*Kpad32
  unsigned* ldsX = (unsigned*)(smem + (size_t)CoutP * Kpad32 * 2);  // CinP*(W+4)

  const int bp = blockIdx.x;
  const int b = bp / P, p = bp % P;
  const int lane = threadIdx.x & 31;
  const int wid  = threadIdx.x >> 5;
  const int nw   = blockDim.x >> 5;
  const int stride = W + 4;

  // ---- stage weights, layout k = ci*4 + t (t==3 and ci>=Cin and co>=Cout: 0)
  for (int co = wid; co < CoutP; co += nw)
    for (int k = lane; k < Kpad32; k += 32) {
      int ci = k >> 2, t = k & 3;
      float v = (co < Cout && t < 3 && ci < Cin)
                    ? wgt[((long)co * Cin + ci) * 3 + t] : 0.f;
      ldsW[co * Kpad32 + k] = (_Float16)v;
    }

  // ---- stage input row: ldsX[ci][j] = half2(Xpad(j), Xpad(j+1)),
  //      Xpad(j) = (1 <= j <= W) ? x[j-1] : 0
  const float* rowB = in + (long)b * Cin * P * W + (long)p * W;
  for (int ci = wid; ci < CinP; ci += nw) {
    const float* row = rowB + (long)ci * P * W;
    for (int j = lane; j < stride; j += 32) {
      float a = 0.f, c = 0.f;
      if (ci < Cin) {
        if (j >= 1 && j <= W) a = row[j - 1];
        if (j < W)            c = row[j];
      }
      union { unsigned u; _Float16 h[2]; } pk;
      pk.h[0] = (_Float16)a;
      pk.h[1] = (_Float16)c;
      ldsX[ci * stride + j] = pk.u;
    }
  }
  __syncthreads();

  // ---- compute tiles
  const int m  = lane & 15;
  const int hh = lane >> 4;
  const int wTiles = W >> 4, cTiles = CoutP >> 4;
  for (int tile = wid; tile < wTiles * cTiles; tile += nw) {
    const int w0  = (tile % wTiles) << 4;
    const int co0 = (tile / wTiles) << 4;
    v8f acc = {0.f, 0.f, 0.f, 0.f, 0.f, 0.f, 0.f, 0.f};
    const _Float16* wr0 = ldsW + (co0 + m) * Kpad32 + hh * 8;
    const unsigned* xr0 = ldsX + w0 + m;
    for (int kk = 0; kk < Kpad32; kk += 32) {
      union { v16h h; uint4 q[2]; } A;
      union { v16h h; unsigned u[8]; } Bf;
      A.q[0] = *(const uint4*)(wr0 + kk);       // k = kk+8*hh   .. +7
      A.q[1] = *(const uint4*)(wr0 + kk + 16);  // k = kk+16+8*hh.. +7
      const int ciBase = (kk >> 2) + hh * 2;
#pragma unroll
      for (int v = 0; v < 8; ++v) {
        int ci = ciBase + (v >> 2) * 4 + ((v & 3) >> 1);
        int tp = (v & 1) * 2;  // tap pair base: halves (t, t+1)
        Bf.u[v] = xr0[ci * stride + tp];
      }
      acc = __builtin_amdgcn_wmma_f32_16x16x32_f16(
          /*neg_a=*/false, A.h, /*neg_b=*/false, Bf.h,
          /*c_mod=*/(short)0, acc, /*reuse_a=*/false, /*reuse_b=*/false);
    }
    const int w = w0 + m;
#pragma unroll
    for (int r = 0; r < 8; ++r) {
      int co = co0 + r + 8 * hh;
      if (co < Cout)
        out[(((long)b * Cout + co) * P + p) * W + w] = acc[r] + bias[co];
    }
  }
}

// relu + width-pool one concat piece: dst[:, coff+c] = relu(max over pairs)
__global__ void k_pool_piece(const float* __restrict__ src, float* __restrict__ dst,
                             int B, int Cp, int P, int Wo, int Ctot, int coff) {
  long total = (long)B * Cp * P * Wo;
  for (long i = (long)blockIdx.x * blockDim.x + threadIdx.x; i < total;
       i += (long)gridDim.x * blockDim.x) {
    int w = (int)(i % Wo);
    long r = i / Wo;
    int p = (int)(r % P); r /= P;
    int c = (int)(r % Cp);
    int b = (int)(r / Cp);
    const float* s = src + (((long)b * Cp + c) * P + p) * (2L * Wo) + 2L * w;
    dst[(((long)b * Ctot + coff + c) * P + p) * Wo + w] =
        fmaxf(fmaxf(s[0], s[1]), 0.f);
  }
}

// upsample2: out[b,o,p,2w+l] = sum_i x[b,i,p,w]*W[i,o,l] + bias[o]; 16 out ch
__global__ void k_upsample(const float* __restrict__ x, const float* __restrict__ w,
                           const float* __restrict__ bias, float* __restrict__ out,
                           int B, int Cin, int P, int Win) {
  int Wout = Win * 2;
  long total = (long)B * 16 * P * Wout;
  for (long i = (long)blockIdx.x * blockDim.x + threadIdx.x; i < total;
       i += (long)gridDim.x * blockDim.x) {
    int wo = (int)(i % Wout);
    long r = i / Wout;
    int p = (int)(r % P); r /= P;
    int o = (int)(r % 16);
    int b = (int)(r / 16);
    int wi = wo >> 1, l = wo & 1;
    float acc = bias[o];
    for (int ci = 0; ci < Cin; ++ci)
      acc += x[(((long)b * Cin + ci) * P + p) * Win + wi] * w[(ci * 16 + o) * 2 + l];
    out[i] = acc;
  }
}

// batchnorm stats: one block per channel, mean/var over (B, P*W)
__global__ __launch_bounds__(256)
void k_bn_stats(const float* __restrict__ x, float* __restrict__ mv,
                int B, int C, int PW) {
  int c = blockIdx.x;
  __shared__ float s1[256], s2[256];
  float a = 0.f, q = 0.f;
  long n = (long)B * PW;
  for (long i = threadIdx.x; i < n; i += blockDim.x) {
    long b = i / PW, r = i % PW;
    float v = x[(b * C + c) * (long)PW + r];
    a += v;
    q += v * v;
  }
  s1[threadIdx.x] = a;
  s2[threadIdx.x] = q;
  __syncthreads();
  for (int s = 128; s > 0; s >>= 1) {
    if ((int)threadIdx.x < s) {
      s1[threadIdx.x] += s1[threadIdx.x + s];
      s2[threadIdx.x] += s2[threadIdx.x + s];
    }
    __syncthreads();
  }
  if (threadIdx.x == 0) {
    float mu = s1[0] / (float)n;
    mv[2 * c]     = mu;
    mv[2 * c + 1] = s2[0] / (float)n - mu * mu;
  }
}

__global__ void k_bn_apply(float* __restrict__ x, const float* __restrict__ mv,
                           const float* __restrict__ g, const float* __restrict__ bb,
                           int C, long PW, long total) {
  for (long i = (long)blockIdx.x * blockDim.x + threadIdx.x; i < total;
       i += (long)gridDim.x * blockDim.x) {
    long c = (i / PW) % C;
    x[i] = (x[i] - mv[2 * c]) * rsqrtf(mv[2 * c + 1] + BN_EPS_F) * g[c] + bb[c];
  }
}

// channel-slot copy for concat
__global__ void k_copy_into(const float* __restrict__ src, float* __restrict__ dst,
                            int B, int C, long PW, int Ctot, int coff) {
  long total = (long)B * C * PW;
  for (long i = (long)blockIdx.x * blockDim.x + threadIdx.x; i < total;
       i += (long)gridDim.x * blockDim.x) {
    long pw = i % PW;
    long r  = i / PW;
    int  c  = (int)(r % C);
    int  b  = (int)(r / C);
    dst[((long)b * Ctot + coff + c) * PW + pw] = src[i];
  }
}

// out[b,hp,w] = sum_c h[b,c,150+hp,w]*ow[c] + ob   (h = (8,17,300,128))
__global__ void k_final(const float* __restrict__ h, const float* __restrict__ ow,
                        const float* __restrict__ ob, float* __restrict__ out) {
  long total = 8L * 150 * 128;
  for (long i = (long)blockIdx.x * blockDim.x + threadIdx.x; i < total;
       i += (long)gridDim.x * blockDim.x) {
    int w  = (int)(i % 128);
    long r = i / 128;
    int hp = (int)(r % 150);
    int b  = (int)(r / 150);
    float acc = ob[0];
    for (int c = 0; c < 17; ++c)
      acc += h[(((long)b * 17 + c) * 300 + 150 + hp) * 128L + w] * ow[c];
    out[i] = acc;
  }
}

// ---------------------------------------------------------------- host side
static inline dim3 gblocks(long n) {
  long g = (n + 255) / 256;
  if (g < 1) g = 1;
  if (g > 1048576) g = 1048576;
  return dim3((unsigned)g);
}

extern "C" void kernel_launch(void* const* d_in, const int* in_sizes, int n_in,
                              void* d_out, int out_size, void* d_ws, size_t ws_size,
                              hipStream_t stream) {
  const int B = 8, P = 300;
  const float* x_in    = (const float*)d_in[0];
  const float* stem_w  = (const float*)d_in[1];
  const float* stem_b  = (const float*)d_in[2];
  const float* stem_lw = (const float*)d_in[3];
  const float* stem_lb = (const float*)d_in[4];
  const float* scalep  = (const float*)d_in[5];
  const float* up_w[3] = {(const float*)d_in[42], (const float*)d_in[43], (const float*)d_in[44]};
  const float* up_b[3] = {(const float*)d_in[45], (const float*)d_in[46], (const float*)d_in[47]};
  const float* bn_g[3] = {(const float*)d_in[48], (const float*)d_in[49], (const float*)d_in[50]};
  const float* bn_b[3] = {(const float*)d_in[51], (const float*)d_in[52], (const float*)d_in[53]};
  const float* out_w   = (const float*)d_in[54];
  const float* out_b   = (const float*)d_in[55];
  const int*   ei      = (const int*)d_in[56];
  const int    E       = (n_in > 56) ? in_sizes[56] / 2 : 153600;
  float* out = (float*)d_out;
  (void)out_size;

  // ---- workspace arena (floats) ----
  size_t need = 0;
  float* base = (float*)d_ws;
  auto alloc = [&](long n) { float* p = base + need; need += (size_t)n; return p; };

  float* dinv = alloc(2400);            // degree -> dinv (nodes < 2400 only)
  float* agg  = alloc(2400L * 128);     // GCN aggregation scratch (max S=128)
  float* g0   = alloc((long)B * 1   * P * 128);  // stem+gcn, skip for k=2
  float* g1   = alloc((long)B * 192 * P * 64);   // blk0 out, skip for k=1
  float* g2   = alloc((long)B * 96  * P * 32);   // blk1 out, skip for k=0
  float* g3   = alloc((long)B * 48  * P * 16);   // blk2 out
  float* bnmv = alloc(32);                       // 16x (mean,var)
  float* scr  = alloc(50700000L);                // shared scratch region
  if (need * sizeof(float) > ws_size) return;    // insufficient scratch

  // ---- degree / dinv (edge set is constant across all 7 GCN calls) ----
  k_zero<<<gblocks(2400), 256, 0, stream>>>(dinv, 2400);
  k_deg<<<gblocks(E), 256, 0, stream>>>(ei, dinv, E);
  k_dinv<<<gblocks(2400), 256, 0, stream>>>(dinv, 2400);

  auto conv = [&](const float* in, const float* w, const float* bias, float* o,
                  int Cin, int Cout, int W) {
    int Kpad32 = ((Cin * 4) + 31) & ~31;   // k = ci*4 + t, rounded to 32
    int CinP   = Kpad32 >> 2;
    int CoutP  = (Cout + 15) & ~15;
    size_t smem = (size_t)CoutP * Kpad32 * 2 + (size_t)CinP * (W + 4) * 4;
    k_conv1x3_wmma<<<dim3((unsigned)(B * P)), 256, smem, stream>>>(
        in, w, bias, o, Cin, Cout, P, W, Kpad32, CinP, CoutP);
  };
  auto lnorm = [&](float* y, const float* lw, const float* lb, const float* res,
                   int C, int gs) {
    k_layernorm<<<dim3((unsigned)(B * C)), 256, 0, stream>>>(y, lw, lb, res, gs);
  };
  auto run_gcn = [&](float* xx, long n, int S) {   // in-place on xx
    k_zero<<<gblocks(2400L * S), 256, 0, stream>>>(agg, 2400L * S);
    k_gcn_scatter<<<gblocks((long)E * S), 256, 0, stream>>>(xx, ei, dinv, agg, E, S);
    k_gcn_finalize<<<dim3(300), 256, 0, stream>>>(xx, agg, scalep, S, 2400);
    if (n > 2400)  // agg==0 for nodes >= 2400 -> output exactly zero
      k_zero<<<gblocks((n - 2400L) * S), 256, 0, stream>>>(xx + 2400L * S,
                                                           (n - 2400L) * S);
  };

  // ---- stem ----
  conv(x_in, stem_w, stem_b, g0, 2, 1, 128);
  lnorm(g0, stem_lw, stem_lb, nullptr, 1, 300 * 128);
  run_gcn(g0, 2400, 128);

  // ---- down res-blocks ----
  struct DP { int cin, cout, W, pb; };
  const DP dps[3] = {{1, 64, 128, 6}, {192, 32, 64, 18}, {96, 16, 32, 30}};
  float* gchain[4] = {g0, g1, g2, g3};
  for (int bi = 0; bi < 3; ++bi) {
    const DP& d = dps[bi];
    const float* cw[3] = {(const float*)d_in[d.pb + 0], (const float*)d_in[d.pb + 1], (const float*)d_in[d.pb + 2]};
    const float* cb[3] = {(const float*)d_in[d.pb + 3], (const float*)d_in[d.pb + 4], (const float*)d_in[d.pb + 5]};
    const float* lw[3] = {(const float*)d_in[d.pb + 6], (const float*)d_in[d.pb + 7], (const float*)d_in[d.pb + 8]};
    const float* lb[3] = {(const float*)d_in[d.pb + 9], (const float*)d_in[d.pb + 10], (const float*)d_in[d.pb + 11]};
    const float* gin = gchain[bi];
    float* gout = gchain[bi + 1];
    float* bufA = scr;
    float* bufB = scr + 19660800L;
    int W = d.W, Wo = W / 2, Cout = d.cout, Ctot = 3 * Cout;
    long pieceN = (long)B * Cout * P * Wo;
    // x1 = ln(conv(in)); pool piece 0
    conv(gin, cw[0], cb[0], bufA, d.cin, Cout, W);
    lnorm(bufA, lw[0], lb[0], nullptr, Cout, P * W);
    k_pool_piece<<<gblocks(pieceN), 256, 0, stream>>>(bufA, gout, B, Cout, P, Wo, Ctot, 0);
    // x2 = ln(conv(x1)) + x1; pool piece 1
    conv(bufA, cw[1], cb[1], bufB, Cout, Cout, W);
    lnorm(bufB, lw[1], lb[1], bufA, Cout, P * W);
    k_pool_piece<<<gblocks(pieceN), 256, 0, stream>>>(bufB, gout, B, Cout, P, Wo, Ctot, Cout);
    // x3 = ln(conv(x2)) + x2; pool piece 2 (reuse bufA)
    conv(bufB, cw[2], cb[2], bufA, Cout, Cout, W);
    lnorm(bufA, lw[2], lb[2], bufB, Cout, P * W);
    k_pool_piece<<<gblocks(pieceN), 256, 0, stream>>>(bufA, gout, B, Cout, P, Wo, Ctot, 2 * Cout);
    run_gcn(gout, (long)B * Ctot * P, Wo);
  }

  // ---- up path ----
  float* ub = scr;                  // upsample buffer (max 8*16*300*128)
  float* h0 = scr + 4915200L;       // (8,112,300,32)
  float* h1 = h0 + 8601600L;        // (8,208,300,64)
  float* h2 = h1 + 31948800L;       // (8, 17,300,128)
  float* houts[3]       = {h0, h1, h2};
  const float* skips[3] = {g2, g1, g0};
  const int    skipC[3] = {96, 192, 1};
  const float* hin = g3;
  int cinU = 48, Win = 16;
  for (int k = 0; k < 3; ++k) {
    int Wout = Win * 2;
    long uN = (long)B * 16 * P * Wout;
    k_upsample<<<gblocks(uN), 256, 0, stream>>>(hin, up_w[k], up_b[k], ub, B, cinU, P, Win);
    k_bn_stats<<<dim3(16), 256, 0, stream>>>(ub, bnmv, B, 16, P * Wout);
    k_bn_apply<<<gblocks(uN), 256, 0, stream>>>(ub, bnmv, bn_g[k], bn_b[k], 16,
                                                (long)P * Wout, uN);
    int Ctot = 16 + skipC[k];
    float* hout = houts[k];
    long PW = (long)P * Wout;
    k_copy_into<<<gblocks(uN), 256, 0, stream>>>(ub, hout, B, 16, PW, Ctot, 0);
    long sN = (long)B * skipC[k] * PW;
    k_copy_into<<<gblocks(sN), 256, 0, stream>>>(skips[k], hout, B, skipC[k], PW, Ctot, 16);
    run_gcn(hout, (long)B * Ctot * P, Wout);
    hin = hout;
    cinU = Ctot;
    Win = Wout;
  }

  // ---- crop + 1x1 output projection ----
  k_final<<<gblocks(8L * 150 * 128), 256, 0, stream>>>(h2, out_w, out_b, out);
}
